// FeatureExtractor_3865470567208
// MI455X (gfx1250) — matrix operand order repaired and checked by reading, not compiled, not run on hardware
//
#include <hip/hip_runtime.h>
#include <stdint.h>

#define DDIM 64

typedef __attribute__((ext_vector_type(16))) __bf16    v16bf;
typedef __attribute__((ext_vector_type(8)))  float     v8f;
typedef __attribute__((ext_vector_type(4)))  uint32_t  u32x4;

union FragBF { v16bf v; u32x4 q[2]; };
union AccF   { v8f  v; float f[8]; };

__device__ __forceinline__ uint16_t f2bf(float f) {
  uint32_t u = __float_as_uint(f);
  uint32_t r = u + 0x7FFFu + ((u >> 16) & 1u);   // round-to-nearest-even
  return (uint16_t)(r >> 16);
}
__device__ __forceinline__ uint32_t f2ord(float f) {   // monotone float -> uint
  uint32_t u = __float_as_uint(f);
  return (u & 0x80000000u) ? ~u : (u | 0x80000000u);
}
__device__ __forceinline__ float ord2f(uint32_t u) {
  uint32_t b = (u & 0x80000000u) ? (u & 0x7FFFFFFFu) : ~u;
  return __uint_as_float(b);
}

// ---------------- utility kernels ----------------

__global__ void k_fill_u32(uint32_t* p, uint32_t v, long long n) {
  long long i  = (long long)blockIdx.x * blockDim.x + threadIdx.x;
  long long st = (long long)gridDim.x * blockDim.x;
  for (; i < n; i += st) p[i] = v;
}

// Pack fp32 W[K,64] into the WMMA B-fragment lane layout (bf16), zero-padded to Kpad.
// Linear index t = ((kt*4 + nblk)*32 + lane); each thread emits 16 contiguous bf16.
__global__ void k_pack_w(const float* __restrict__ W, int K, int Kpad,
                         uint16_t* __restrict__ Wp) {
  int t = blockIdx.x * blockDim.x + threadIdx.x;
  int total = (Kpad >> 5) * 128;
  if (t >= total) return;
  int lane = t & 31, n = (t >> 5) & 3, kt = t >> 7;
  int col = n * 16 + (lane & 15);
  int kb = kt * 32 + ((lane & 16) ? 8 : 0);
  uint16_t* dst = Wp + (size_t)t * 16;
#pragma unroll
  for (int j = 0; j < 16; ++j) {
    int k = kb + j + ((j & 8) ? 8 : 0);
    dst[j] = (k < K) ? f2bf(W[(size_t)k * DDIM + col]) : (uint16_t)0;
  }
}

// concat(p0[M,K0], p1[M,K1]) -> bf16 [Mpad, Kpad] (row-major, zero pad).
// p0 row stride == K0 (our feature buffers use stride 64 == K0 there).
__global__ void k_build_comb(const float* __restrict__ p0, int K0,
                             const float* __restrict__ p1, int K1,
                             uint16_t* __restrict__ dst, int Kpad,
                             int M, long long total) {
  long long t = (long long)blockIdx.x * blockDim.x + threadIdx.x;
  if (t >= total) return;
  int col = (int)(t % Kpad);
  long long row = t / Kpad;
  float f = 0.f;
  if (row < M) {
    if (col < K0) f = p0[row * (long long)K0 + col];
    else if (col < K0 + K1) f = p1[row * (long long)K1 + (col - K0)];
  }
  dst[t] = f2bf(f);
}

// edge_comb = concat(edge_f[:, :K0] (stride 64), lo, hi, def_mm, edge_lp[NE,8]) -> bf16
__global__ void k_build_ecomb(const float* __restrict__ ef, int K0,
                              const float* __restrict__ lo, const float* __restrict__ hi,
                              const float* __restrict__ dm, const float* __restrict__ elp,
                              uint16_t* __restrict__ dst, int Kpad, int M, long long total) {
  long long t = (long long)blockIdx.x * blockDim.x + threadIdx.x;
  if (t >= total) return;
  int col = (int)(t % Kpad);
  long long row = t / Kpad;
  float f = 0.f;
  if (row < M) {
    if (col < K0) f = ef[row * 64 + col];
    else {
      int c = col - K0;
      if (c == 0) f = lo[row];
      else if (c == 1) f = hi[row];
      else if (c == 2) f = dm[row];
      else if (c < 11) f = elp[row * 8 + (c - 3)];
    }
  }
  dst[t] = f2bf(f);
}

// cat3 = concat(ecomb[:, :ed] (bf16, stride KpadE), vc[var_idx], cc[con_idx]) -> bf16
__global__ void k_build_cat3(const uint16_t* __restrict__ ecomb, int KpadE, int ed,
                             const float* __restrict__ vc, const int* __restrict__ vidx,
                             const float* __restrict__ cc, const int* __restrict__ cidx,
                             uint16_t* __restrict__ dst, int Kpad, int M, long long total) {
  long long t = (long long)blockIdx.x * blockDim.x + threadIdx.x;
  if (t >= total) return;
  int col = (int)(t % Kpad);
  long long row = t / Kpad;
  uint16_t o = 0;
  if (row < M) {
    if (col < ed) o = ecomb[row * (long long)KpadE + col];
    else if (col < ed + 64) {
      int v = vidx[row];
      o = f2bf(vc[(size_t)v * DDIM + (col - ed)]);
    } else if (col < ed + 128) {
      int c = cidx[row];
      o = f2bf(cc[(size_t)c * DDIM + (col - ed - 64)]);
    }
  }
  dst[t] = o;
}

// ---------------- WMMA GEMM: C[Mpad,64] = (relu)(A[Mpad,Kpad]_bf16 @ Wpacked + bias) ----
// 256 threads = 8 waves; each wave computes a 16x64 tile with 4 f32 accumulators,
// K-loop in chunks of 32 using v_wmma_f32_16x16x32_bf16.
// The packed weight tile (Kpad*128 bytes, shared by all waves) is staged in LDS once
// per block; the four B fragments per K-step are loaded into distinct registers so
// the 8 ds_load_b128 can be clause'd with a single DS wait, followed by 4 WMMAs
// issued back-to-back (disjoint accumulators -> no hazards, good co-execution).
__global__ void __launch_bounds__(256) k_gemm_bf16(
    const uint16_t* __restrict__ A, const uint16_t* __restrict__ Wp,
    const float* __restrict__ bias, float* __restrict__ C,
    int Kpad, int relu, long long Mpad) {
  extern __shared__ uint16_t smem[];   // Kpad*64 bf16 packed fragments

  // cooperative coalesced copy of the weight tile into LDS (16B per thread-step)
  {
    int nvec = Kpad * 8;               // number of u32x4 chunks = Kpad*128/16
    const u32x4* src = reinterpret_cast<const u32x4*>(Wp);
    u32x4* dst = reinterpret_cast<u32x4*>(smem);
    for (int i = threadIdx.x; i < nvec; i += 256) dst[i] = src[i];
  }
  __syncthreads();

  int lane = threadIdx.x & 31;
  long long rowbase = ((long long)blockIdx.x * 8 + (threadIdx.x >> 5)) * 16;
  AccF acc[4];
#pragma unroll
  for (int n = 0; n < 4; ++n)
#pragma unroll
    for (int j = 0; j < 8; ++j) acc[n].f[j] = 0.f;

  const uint16_t* arow =
      A + (rowbase + (lane & 15)) * (size_t)Kpad + ((lane & 16) ? 8 : 0);
  const uint16_t* wl = smem + (size_t)lane * 16;
  int nkt = Kpad >> 5;
  for (int kt = 0; kt < nkt; ++kt) {
    FragBF a;
    a.q[0] = *reinterpret_cast<const u32x4*>(arow);        // K {0..7} / {8..15}
    a.q[1] = *reinterpret_cast<const u32x4*>(arow + 16);   // K {16..23} / {24..31}
    arow += 32;
    const uint16_t* w = wl + (size_t)kt * 2048;
    FragBF b0, b1, b2, b3;                 // distinct regs -> one DS wait per K-step
    b0.q[0] = *reinterpret_cast<const u32x4*>(w);
    b0.q[1] = *reinterpret_cast<const u32x4*>(w + 8);
    b1.q[0] = *reinterpret_cast<const u32x4*>(w + 512);
    b1.q[1] = *reinterpret_cast<const u32x4*>(w + 520);
    b2.q[0] = *reinterpret_cast<const u32x4*>(w + 1024);
    b2.q[1] = *reinterpret_cast<const u32x4*>(w + 1032);
    b3.q[0] = *reinterpret_cast<const u32x4*>(w + 1536);
    b3.q[1] = *reinterpret_cast<const u32x4*>(w + 1544);
    acc[0].v = __builtin_amdgcn_wmma_f32_16x16x32_bf16(
        false, a.v, false, b0.v, (short)0, acc[0].v, false, false);
    acc[1].v = __builtin_amdgcn_wmma_f32_16x16x32_bf16(
        false, a.v, false, b1.v, (short)0, acc[1].v, false, false);
    acc[2].v = __builtin_amdgcn_wmma_f32_16x16x32_bf16(
        false, a.v, false, b2.v, (short)0, acc[2].v, false, false);
    acc[3].v = __builtin_amdgcn_wmma_f32_16x16x32_bf16(
        false, a.v, false, b3.v, (short)0, acc[3].v, false, false);
  }
  int colb = lane & 15;
  long long r0 = rowbase + ((lane & 16) ? 8 : 0);
#pragma unroll
  for (int n = 0; n < 4; ++n) {
    float bs = bias ? bias[n * 16 + colb] : 0.f;
#pragma unroll
    for (int j = 0; j < 8; ++j) {
      float x = acc[n].f[j] + bs;
      if (relu) x = fmaxf(x, 0.f);
      C[(r0 + j) * DDIM + n * 16 + colb] = x;
    }
  }
}

// ---------------- edge attention kernels ----------------

__global__ void k_score(const float* __restrict__ q, const float* __restrict__ kk,
                        const float* __restrict__ e, const int* __restrict__ sidx,
                        const int* __restrict__ didx, float* __restrict__ score, int NEe) {
  int t = blockIdx.x * blockDim.x + threadIdx.x;
  if (t >= NEe) return;
  const float4* qp = reinterpret_cast<const float4*>(q + (size_t)didx[t] * DDIM);
  const float4* kp = reinterpret_cast<const float4*>(kk + (size_t)sidx[t] * DDIM);
  const float4* ep = reinterpret_cast<const float4*>(e + (size_t)t * DDIM);
  float acc = 0.f;
#pragma unroll
  for (int i = 0; i < DDIM / 4; ++i) {
    float4 a = qp[i], b = kp[i], c = ep[i];
    acc += a.x * (b.x + c.x) + a.y * (b.y + c.y) + a.z * (b.z + c.z) + a.w * (b.w + c.w);
  }
  score[t] = acc * 0.125f;   // 1/sqrt(64)
}

__global__ void k_segmax(const float* __restrict__ score, const int* __restrict__ didx,
                         uint32_t* __restrict__ smax, int NEe) {
  int t = blockIdx.x * blockDim.x + threadIdx.x;
  if (t >= NEe) return;
  atomicMax(&smax[didx[t]], f2ord(score[t]));
}

__global__ void k_softnum(const float* __restrict__ score, const int* __restrict__ didx,
                          const uint32_t* __restrict__ smax, float* __restrict__ ex,
                          float* __restrict__ denom, float* __restrict__ deg, int NEe) {
  int t = blockIdx.x * blockDim.x + threadIdx.x;
  if (t >= NEe) return;
  int d = didx[t];
  float m = ord2f(smax[d]);
  if (!isfinite(m)) m = 0.f;
  float v = __expf(score[t] - m);
  ex[t] = v;
  atomicAdd(&denom[d], v);
  atomicAdd(&deg[d], 1.f);
}

// agg[dst] += alpha * (v[src] + e);  4 threads per edge (16 cols each)
__global__ void k_scatter(const float* __restrict__ ex, const float* __restrict__ denom,
                          const float* __restrict__ vv, const float* __restrict__ e,
                          const int* __restrict__ sidx, const int* __restrict__ didx,
                          float* __restrict__ agg, int NEe) {
  int t = blockIdx.x * blockDim.x + threadIdx.x;
  int edge = t >> 2, part = t & 3;
  if (edge >= NEe) return;
  int s = sidx[edge], d = didx[edge];
  float alpha = ex[edge] / (denom[d] + 1e-16f);
  const float4* vp = reinterpret_cast<const float4*>(vv + (size_t)s * DDIM) + part * 4;
  const float4* ep = reinterpret_cast<const float4*>(e + (size_t)edge * DDIM) + part * 4;
  float* ap = agg + (size_t)d * DDIM + part * 16;
#pragma unroll
  for (int i = 0; i < 4; ++i) {
    float4 a = vp[i], b = ep[i];
    atomicAdd(ap + i * 4 + 0, alpha * (a.x + b.x));
    atomicAdd(ap + i * 4 + 1, alpha * (a.y + b.y));
    atomicAdd(ap + i * 4 + 2, alpha * (a.z + b.z));
    atomicAdd(ap + i * 4 + 3, alpha * (a.w + b.w));
  }
}

__global__ void k_finalize(const float* __restrict__ agg, const float* __restrict__ deg,
                           const float* __restrict__ skip, float* __restrict__ out,
                           long long total) {
  long long t = (long long)blockIdx.x * blockDim.x + threadIdx.x;
  if (t >= total) return;
  long long node = t >> 6;
  float dg = fmaxf(deg[node], 1.f);
  out[t] = fmaxf(agg[t] / dg + skip[t], 0.f);
}

// ---------------- host ----------------

extern "C" void kernel_launch(void* const* d_in, const int* in_sizes, int n_in,
                              void* d_out, int out_size, void* d_ws, size_t ws_size,
                              hipStream_t stream) {
  if (n_in < 67) return;
  const int FV = 8, FC = 8;
  const int NV = in_sizes[0] / FV;
  const int NC = in_sizes[1] / FC;
  const int NE = in_sizes[2];

  const float* var_lp = (const float*)d_in[0];
  const float* con_lp = (const float*)d_in[1];
  const float* lo  = (const float*)d_in[2];
  const float* hi  = (const float*)d_in[3];
  const float* dm  = (const float*)d_in[4];
  const float* elp = (const float*)d_in[5];
  const int* eidx  = (const int*)d_in[6];       // jax default int32
  const int* vidx = eidx;
  const int* cidx = eidx + NE;
  auto F = [&](int i) { return (const float*)d_in[i]; };

  auto ruL = [](long long x, long long a) { return (x + a - 1) / a * a; };
  const long long NVp = ruL(NV, 128), NCp = ruL(NC, 128), NEp = ruL(NE, 128);

  // ---- workspace layout (deterministic bump allocator) ----
  char* basep = (char*)d_ws;
  size_t off = 0;
  auto alloc = [&](size_t bytes) -> void* {
    void* p = basep + off;
    off = (off + bytes + 255) & ~(size_t)255;
    return p;
  };
  float*    var_f   = (float*)alloc(NVp * 64 * 4);
  float*    con_f   = (float*)alloc(NCp * 64 * 4);
  float*    edge_f  = (float*)alloc(NEp * 64 * 4);
  uint16_t* vcomb   = (uint16_t*)alloc(NVp * 96 * 2);
  uint16_t* vcomb2  = (uint16_t*)alloc(NVp * 96 * 2);
  uint16_t* ccomb   = (uint16_t*)alloc(NCp * 96 * 2);
  uint16_t* ccomb2  = (uint16_t*)alloc(NCp * 96 * 2);
  uint16_t* ecomb   = (uint16_t*)alloc(NEp * 96 * 2);
  uint16_t* catbuf  = (uint16_t*)alloc(NEp * 224 * 2);
  float*    eattr   = (float*)alloc(NEp * 64 * 4);   // edge lin + reused as edge-mlp hidden
  uint16_t* ehid_bf = (uint16_t*)alloc(NEp * 64 * 2);
  float*    exbuf   = (float*)alloc(NEp * 4);        // score, then exp
  float*    tq      = (float*)alloc(NVp * 64 * 4);
  float*    tk      = (float*)alloc(NVp * 64 * 4);
  float*    tv      = (float*)alloc(NVp * 64 * 4);
  float*    tskip   = (float*)alloc(NVp * 64 * 4);
  float*    tagg    = (float*)alloc(NVp * 64 * 4);
  uint32_t* smax    = (uint32_t*)alloc(NVp * 4);
  float*    denom   = (float*)alloc(NVp * 4);
  float*    deg     = (float*)alloc(NVp * 4);
  float*    vc1     = (float*)alloc(NVp * 64 * 4);
  float*    vc2     = (float*)alloc(NVp * 64 * 4);
  uint16_t* vc1_bf  = (uint16_t*)alloc(NVp * 64 * 2);
  float*    cc1     = (float*)alloc(NCp * 64 * 4);
  float*    cc2     = (float*)alloc(NCp * 64 * 4);
  uint16_t* cc1_bf  = (uint16_t*)alloc(NCp * 64 * 2);
  uint16_t* wpk     = (uint16_t*)alloc((size_t)4096 * 64 * 2);  // packed weights
  size_t wpoff = 0;

  auto pack = [&](const float* W, int K) -> const uint16_t* {
    int Kpad = (int)ruL(K, 32);
    uint16_t* p = wpk + wpoff;
    wpoff += (size_t)Kpad * 64;
    int total = (Kpad / 32) * 128;
    k_pack_w<<<(total + 127) / 128, 128, 0, stream>>>(W, K, Kpad, p);
    return p;
  };
  auto gemm = [&](const uint16_t* A, long long Mpad, int Kpad, const uint16_t* Wp,
                  const float* bias, float* C, int relu) {
    size_t shbytes = (size_t)Kpad * 128;   // Kpad*64 bf16 in LDS
    k_gemm_bf16<<<(int)(Mpad / 128), 256, shbytes, stream>>>(A, Wp, bias, C, Kpad,
                                                             relu, Mpad);
  };
  auto fillu = [&](void* p, uint32_t v, long long n) {
    int blocks = (int)((n + 255) / 256);
    if (blocks > 4096) blocks = 4096;
    k_fill_u32<<<blocks, 256, 0, stream>>>((uint32_t*)p, v, n);
  };

  // TransformerConv: src features -> dst features (writes outF = relu(conv))
  auto tconv = [&](const uint16_t* src_bf, int Kps, long long Nsp,
                   const uint16_t* dst_bf, int Kpd, long long Ndp, int Nd,
                   const int* sidx, const int* didx,
                   const uint16_t* pk_edge,
                   const uint16_t* pk_key, const float* b_key,
                   const uint16_t* pk_qry, const float* b_qry,
                   const uint16_t* pk_skip, const float* b_skip,
                   const uint16_t* pk_val, const float* b_val,
                   int Kpe, float* outF) {
    gemm(dst_bf, Ndp, Kpd, pk_qry, b_qry, tq, 0);
    gemm(src_bf, Nsp, Kps, pk_key, b_key, tk, 0);
    gemm(src_bf, Nsp, Kps, pk_val, b_val, tv, 0);
    gemm(dst_bf, Ndp, Kpd, pk_skip, b_skip, tskip, 0);
    gemm(ecomb, NEp, Kpe, pk_edge, nullptr, eattr, 0);
    fillu(smax, 0x007FFFFFu, Nd);            // f2ord(-inf)
    fillu(denom, 0u, Nd);
    fillu(deg, 0u, Nd);
    fillu(tagg, 0u, (long long)Nd * 64);
    int gb = (NE + 255) / 256;
    k_score<<<gb, 256, 0, stream>>>(tq, tk, eattr, sidx, didx, exbuf, NE);
    k_segmax<<<gb, 256, 0, stream>>>(exbuf, didx, smax, NE);
    k_softnum<<<gb, 256, 0, stream>>>(exbuf, didx, smax, exbuf, denom, deg, NE);
    k_scatter<<<(NE * 4 + 255) / 256, 256, 0, stream>>>(exbuf, denom, tv, eattr,
                                                        sidx, didx, tagg, NE);
    long long tot = (long long)Nd * 64;
    k_finalize<<<(int)((tot + 255) / 256), 256, 0, stream>>>(tagg, deg, tskip, outF, tot);
  };

  for (int l = 0; l < 2; ++l) {
    int iv = (l == 0) ? 0 : 64;
    int vd = FV + iv, cd = FC + iv, ed = 11 + iv;
    int Kpv = (int)ruL(vd, 32), Kpc = (int)ruL(cd, 32), Kpe = (int)ruL(ed, 32);
    int Kpcat = (int)ruL(ed + 128, 32);
    int pb = 7 + l * 30;   // pytree leaf order: con_conv, edge_upd, var_conv

    const uint16_t* cc_edge = pack(F(pb + 0), ed);
    const uint16_t* cc_key  = pack(F(pb + 1), vd);
    const uint16_t* cc_qry  = pack(F(pb + 3), cd);
    const uint16_t* cc_skip = pack(F(pb + 5), cd);
    const uint16_t* cc_val  = pack(F(pb + 7), vd);
    const uint16_t* cm_W0   = pack(F(pb + 9), 72);
    const uint16_t* cm_W1   = pack(F(pb + 11), 64);
    const uint16_t* em_W0   = pack(F(pb + 13), ed + 128);
    const uint16_t* em_W1   = pack(F(pb + 15), 64);
    const uint16_t* vm_W0   = pack(F(pb + 17), 72);
    const uint16_t* vm_W1   = pack(F(pb + 19), 64);
    const uint16_t* vv_edge = pack(F(pb + 21), ed);
    const uint16_t* vv_key  = pack(F(pb + 22), 72);
    const uint16_t* vv_qry  = pack(F(pb + 24), vd);
    const uint16_t* vv_skip = pack(F(pb + 26), vd);
    const uint16_t* vv_val  = pack(F(pb + 28), 72);

    long long t;
    t = NVp * Kpv;
    k_build_comb<<<(int)((t + 255) / 256), 256, 0, stream>>>(
        l ? var_f : nullptr, l ? 64 : 0, var_lp, FV, vcomb, Kpv, NV, t);
    t = NCp * Kpc;
    k_build_comb<<<(int)((t + 255) / 256), 256, 0, stream>>>(
        l ? con_f : nullptr, l ? 64 : 0, con_lp, FC, ccomb, Kpc, NC, t);
    t = NEp * Kpe;
    k_build_ecomb<<<(int)((t + 255) / 256), 256, 0, stream>>>(
        edge_f, l ? 64 : 0, lo, hi, dm, elp, ecomb, Kpe, NE, t);

    // con_conv: var -> con
    tconv(vcomb, Kpv, NVp, ccomb, Kpc, NCp, NC, vidx, cidx,
          cc_edge, cc_key, F(pb + 2), cc_qry, F(pb + 4),
          cc_skip, F(pb + 6), cc_val, F(pb + 8), Kpe, con_f);

    t = NCp * 96;
    k_build_comb<<<(int)((t + 255) / 256), 256, 0, stream>>>(
        con_f, 64, con_lp, FC, ccomb2, 96, NC, t);

    // var_conv: con -> var (dst features are pre-update var_comb)
    tconv(ccomb2, 96, NCp, vcomb, Kpv, NVp, NV, cidx, vidx,
          vv_edge, vv_key, F(pb + 23), vv_qry, F(pb + 25),
          vv_skip, F(pb + 27), vv_val, F(pb + 29), Kpe, var_f);

    t = NVp * 96;
    k_build_comb<<<(int)((t + 255) / 256), 256, 0, stream>>>(
        var_f, 64, var_lp, FV, vcomb2, 96, NV, t);

    // edge update
    gemm(vcomb2, NVp, 96, vm_W0, F(pb + 18), vc1, 1);
    t = NVp * 64;
    k_build_comb<<<(int)((t + 255) / 256), 256, 0, stream>>>(
        nullptr, 0, vc1, 64, vc1_bf, 64, NV, t);
    gemm(vc1_bf, NVp, 64, vm_W1, F(pb + 20), vc2, 1);

    gemm(ccomb2, NCp, 96, cm_W0, F(pb + 10), cc1, 1);
    t = NCp * 64;
    k_build_comb<<<(int)((t + 255) / 256), 256, 0, stream>>>(
        nullptr, 0, cc1, 64, cc1_bf, 64, NC, t);
    gemm(cc1_bf, NCp, 64, cm_W1, F(pb + 12), cc2, 1);

    t = NEp * Kpcat;
    k_build_cat3<<<(int)((t + 255) / 256), 256, 0, stream>>>(
        ecomb, Kpe, ed, vc2, vidx, cc2, cidx, catbuf, Kpcat, NE, t);
    gemm(catbuf, NEp, Kpcat, em_W0, F(pb + 14), eattr, 1);   // eattr reused as hidden
    t = NEp * 64;
    k_build_comb<<<(int)((t + 255) / 256), 256, 0, stream>>>(
        nullptr, 0, eattr, 64, ehid_bf, 64, NE, t);
    gemm(ehid_bf, NEp, 64, em_W1, F(pb + 16), edge_f, 1);
  }

  float* out = (float*)d_out;
  hipMemcpyAsync(out, var_f, (size_t)NV * 64 * 4, hipMemcpyDeviceToDevice, stream);
  hipMemcpyAsync(out + (size_t)NV * 64, con_f, (size_t)NC * 64 * 4,
                 hipMemcpyDeviceToDevice, stream);
  hipMemcpyAsync(out + (size_t)(NV + NC) * 64, edge_f, (size_t)NE * 64 * 4,
                 hipMemcpyDeviceToDevice, stream);
}
